// QLinearLayerV2_85761906967213
// MI455X (gfx1250) — 8-bit matrix operand order repaired and checked by reading, compiled, not run on hardware
//
#include <hip/hip_runtime.h>
#include <stdint.h>


typedef __attribute__((ext_vector_type(8))) int   v8i;
typedef __attribute__((ext_vector_type(4))) int   v4i;
typedef __attribute__((ext_vector_type(2))) int   v2i;
typedef __attribute__((ext_vector_type(4))) float v4f;

static constexpr int KDIM = 4096;
static constexpr int BLK  = 128;
static constexpr int CBLK = KDIM / BLK;

__global__ __launch_bounds__(256) void quant_rows_kernel(
    const float* __restrict__ src,
    int8_t* __restrict__ q,
    float* __restrict__ scales_t,
    int rows)
{
    const int lane = threadIdx.x & 31;
    const int gw   = (int)((blockIdx.x * 256u + threadIdx.x) >> 5);
    if (gw >= rows * CBLK) return;
    const int row = gw / CBLK;
    const int c   = gw % CBLK;

    const size_t base = (size_t)row * KDIM + (size_t)c * BLK;
    const v4f v = *reinterpret_cast<const v4f*>(src + base + lane * 4);

    float m = fmaxf(fmaxf(fabsf(v[0]), fabsf(v[1])),
                    fmaxf(fabsf(v[2]), fabsf(v[3])));
#pragma unroll
    for (int off = 16; off > 0; off >>= 1)
        m = fmaxf(m, __shfl_xor(m, off, 32));

    float s = m * (1.0f / 127.0f);
    if (s == 0.0f) s = 1.0f;
    const float inv = 1.0f / s;

    const int q0 = (int)rintf(v[0] * inv);
    const int q1 = (int)rintf(v[1] * inv);
    const int q2 = (int)rintf(v[2] * inv);
    const int q3 = (int)rintf(v[3] * inv);
    const int packed = (q0 & 0xff) | ((q1 & 0xff) << 8) |
                       ((q2 & 0xff) << 16) | (q3 << 24);
    *reinterpret_cast<int*>(q + base + lane * 4) = packed;

    if (lane == 0) scales_t[(size_t)c * rows + row] = s;
}

__device__ __forceinline__ v8i load_a_frag(const int8_t* p  ) {
    const v2i t0 = *reinterpret_cast<const v2i*>(p);
    const v2i t1 = *reinterpret_cast<const v2i*>(p + 8);
    const v2i t2 = *reinterpret_cast<const v2i*>(p + 32);
    const v2i t3 = *reinterpret_cast<const v2i*>(p + 40);
    v8i r;
    r[0] = t0[0]; r[1] = t0[1]; r[2] = t1[0]; r[3] = t1[1];
    r[4] = t2[0]; r[5] = t2[1]; r[6] = t3[0]; r[7] = t3[1];
    return r;
}
__device__ __forceinline__ v8i load_b_frag(const int8_t* p  ) {
    const v4i t0 = *reinterpret_cast<const v4i*>(p);
    const v4i t1 = *reinterpret_cast<const v4i*>(p + 32);
    v8i r;
    r[0] = t0[0]; r[1] = t0[1]; r[2] = t0[2]; r[3] = t0[3];
    r[4] = t1[0]; r[5] = t1[1]; r[6] = t1[2]; r[7] = t1[3];
    return r;
}

__global__ __launch_bounds__(256) void q8_blockscale_gemm_kernel(
    const int8_t* __restrict__ qx,
    const int8_t* __restrict__ qw,
    const float*  __restrict__ xs_t,
    const float*  __restrict__ ws_t,
    float* __restrict__ out,
    int M, int N)
{
    const int lane = threadIdx.x & 31;
    const int w    = threadIdx.x >> 5;
    const int l16  = lane & 15;
    const int kh   = lane >> 4;

    const int row0 = blockIdx.y * 128 + (w >> 1) * 32;
    const int col0 = blockIdx.x * 128 + (w & 1) * 64;

    float accf[2][4][8];
#pragma unroll
    for (int mt = 0; mt < 2; ++mt)
#pragma unroll
        for (int nt = 0; nt < 4; ++nt)
#pragma unroll
            for (int j = 0; j < 8; ++j) accf[mt][nt][j] = 0.0f;

    const int8_t* aP[2];
#pragma unroll
    for (int mt = 0; mt < 2; ++mt)
        aP[mt] = qx + (size_t)(row0 + mt * 16 + l16) * KDIM + kh * 16;
    const int8_t* bP[4];
#pragma unroll
    for (int nt = 0; nt < 4; ++nt)
        bP[nt] = qw + (size_t)(col0 + nt * 16 + l16) * KDIM + kh * 16;

    for (int c = 0; c < CBLK; ++c) {
        const int k0 = c * BLK;

        v4f xsv[2][2];
#pragma unroll
        for (int mt = 0; mt < 2; ++mt) {
            const float* sp = xs_t + (size_t)c * M + row0 + mt * 16 + kh * 8;
            xsv[mt][0] = *reinterpret_cast<const v4f*>(sp);
            xsv[mt][1] = *reinterpret_cast<const v4f*>(sp + 4);
        }
        float wsv[4];
#pragma unroll
        for (int nt = 0; nt < 4; ++nt)
            wsv[nt] = ws_t[(size_t)c * N + col0 + nt * 16 + l16];

        v8i af[2][2], bf[4][2];
#pragma unroll
        for (int mt = 0; mt < 2; ++mt) {
            af[mt][0] = load_a_frag(aP[mt] + k0);
            af[mt][1] = load_a_frag(aP[mt] + k0 + 64);
        }
#pragma unroll
        for (int nt = 0; nt < 4; ++nt) {
            bf[nt][0] = load_b_frag(bP[nt] + k0);
            bf[nt][1] = load_b_frag(bP[nt] + k0 + 64);
        }

        if (c + 1 < CBLK) {
#pragma unroll
            for (int mt = 0; mt < 2; ++mt) __builtin_prefetch(aP[mt] + k0 + BLK, 0, 0);
#pragma unroll
            for (int nt = 0; nt < 4; ++nt) __builtin_prefetch(bP[nt] + k0 + BLK, 0, 0);
        }

#pragma unroll
        for (int mt = 0; mt < 2; ++mt) {
#pragma unroll
            for (int nt = 0; nt < 4; ++nt) {
                v8i p = {};
                p = __builtin_amdgcn_wmma_i32_16x16x64_iu8(
                        true, af[mt][0], true, bf[nt][0], p, false, false);
                p = __builtin_amdgcn_wmma_i32_16x16x64_iu8(
                        true, af[mt][1], true, bf[nt][1], p, false, false);
                const float wsn = wsv[nt];
#pragma unroll
                for (int j = 0; j < 8; ++j) {
                    const float sc = xsv[mt][j >> 2][j & 3] * wsn;
                    accf[mt][nt][j] += (float)p[j] * sc;
                }
            }
        }
    }

#pragma unroll
    for (int mt = 0; mt < 2; ++mt)
#pragma unroll
        for (int nt = 0; nt < 4; ++nt)
#pragma unroll
            for (int j = 0; j < 8; ++j) {
                const int r  = row0 + mt * 16 + kh * 8 + j;
                const int cn = col0 + nt * 16 + l16;
                out[(size_t)r * N + cn] = accf[mt][nt][j];
            }
}

extern "C" void kernel_launch(void* const* d_in, const int* in_sizes, int n_in,
                              void* d_out, int out_size, void* d_ws, size_t ws_size,
                              hipStream_t stream) {
    const float* x      = (const float*)d_in[0];
    const float* weight = (const float*)d_in[1];
    float* out          = (float*)d_out;

    const int M = in_sizes[0] / KDIM;
    const int N = in_sizes[1] / KDIM;

    uint8_t* ws = (uint8_t*)d_ws;
    size_t off = 0;
    auto carve = [&](size_t bytes) -> void* {
        void* p = ws + off;
        off = (off + bytes + 255) & ~(size_t)255;
        return p;
    };
    int8_t* qx  = (int8_t*)carve((size_t)M * KDIM);
    int8_t* qw  = (int8_t*)carve((size_t)N * KDIM);
    float*  xst = (float*)carve((size_t)CBLK * M * sizeof(float));
    float*  wst = (float*)carve((size_t)CBLK * N * sizeof(float));
    (void)ws_size; (void)n_in; (void)out_size;

    {
        int waves  = M * CBLK;
        int blocks = (waves + 7) / 8;
        quant_rows_kernel<<<blocks, 256, 0, stream>>>(x, qx, xst, M);
    }
    {
        int waves  = N * CBLK;
        int blocks = (waves + 7) / 8;
        quant_rows_kernel<<<blocks, 256, 0, stream>>>(weight, qw, wst, N);
    }
    dim3 grid(N / 128, M / 128);
    q8_blockscale_gemm_kernel<<<grid, 256, 0, stream>>>(qx, qw, xst, wst, out, M, N);
}
